// SteerableGeometricProductLayer_83468394430634
// MI455X (gfx1250) — matrix operand order repaired and checked by reading, not compiled, not run on hardware
//
#include <hip/hip_runtime.h>

// ---------------- Problem constants (from reference setup_inputs) ----------------
#define B_TOT 4096
#define F_DIM 512
#define NBB   256   // B/16   (16-row pack blocks)
#define NBB2  128   // B/32   (32-row GEMM tiles)
#define NMB   16    // F/32   (K chunks of 32)
#define NNB   32    // F/16   (16-col output tiles)

typedef __attribute__((ext_vector_type(4)))  int     v4i;
typedef __attribute__((ext_vector_type(8)))  int     v8i;
typedef __attribute__((ext_vector_type(4)))  float   v4f;
typedef __attribute__((ext_vector_type(8)))  float   v8f;
typedef __attribute__((ext_vector_type(16))) __bf16  v16bf;

// ---------------- Compile-time Cl(1,3) tables ----------------
constexpr int kBlade[16] = {0, 1,2,4,8, 3,5,6,9,10,12, 7,11,13,14, 15};

constexpr int cpopc(int x){ int c=0; while(x){ c += x&1; x >>= 1; } return c; }

constexpr int idxOfMask(int m){
  for (int i=0;i<16;i++) if (kBlade[i]==m) return i;
  return 0;
}

constexpr float prodSign(int a,int b){
  int s=0, t=a>>1;
  while(t){ s += cpopc(t&b); t >>= 1; }
  float sg = (s&1) ? -1.0f : 1.0f;
  for (int i=0;i<4;i++)
    if (((a>>i)&1) && ((b>>i)&1)) sg *= (i==0 ? 1.0f : -1.0f);  // metric (+,-,-,-)
  return sg;
}

constexpr bool pathOK(int p,int q,int r){
  int lo = (p>r) ? p-r : r-p;
  int hi = (p+r < 8-(p+r)) ? p+r : 8-(p+r);
  return (((p+q+r)&1)==0) && q>=lo && q<=hi;
}

constexpr int pathIndex(int p,int q,int r){
  int idx=0;
  for(int a=0;a<5;a++) for(int b=0;b<5;b++) for(int c=0;c<5;c++){
    if (pathOK(a,b,c)){ if (a==p && b==q && c==r) return idx; idx++; }
  }
  return 0;
}

struct GPTables {
  int   jj[16][16];   // output component of e_i * e_k
  int   pp[16][16];   // index into gp_weight[.,35]
  float ss[16][16];   // sign of e_i * e_k
  int   grade[16];
  float beta[16];     // blade quadratic-form signs
};

constexpr GPTables makeTables(){
  GPTables t{};
  for (int i=0;i<16;i++){
    t.grade[i] = cpopc(kBlade[i]);
    float bt = 1.0f;
    for (int bb=0; bb<4; bb++)
      if ((kBlade[i]>>bb)&1) bt *= (bb==0 ? 1.0f : -1.0f);
    t.beta[i] = bt;
  }
  for (int i=0;i<16;i++) for (int k=0;k<16;k++){
    int j = idxOfMask(kBlade[i]^kBlade[k]);
    t.jj[i][k] = j;
    t.ss[i][k] = prodSign(kBlade[i], kBlade[k]);
    t.pp[i][k] = pathIndex(t.grade[i], t.grade[j], t.grade[k]);
  }
  return t;
}

constexpr GPTables GT = makeTables();

// ---------------- helpers ----------------
static __device__ inline unsigned pack_bf16(float a, float b){
  __bf16 x = (__bf16)a, y = (__bf16)b;
  unsigned short ux = __builtin_bit_cast(unsigned short, x);
  unsigned short uy = __builtin_bit_cast(unsigned short, y);
  return (unsigned)ux | ((unsigned)uy << 16);
}

static __device__ inline v16bf load_bf16x16(const unsigned* p){
  v4i a = *(const v4i*)p;
  v4i c = *(const v4i*)(p + 4);
  union { v8i i; v16bf v; } u;
  u.i[0]=a[0]; u.i[1]=a[1]; u.i[2]=a[2]; u.i[3]=a[3];
  u.i[4]=c[0]; u.i[5]=c[1]; u.i[6]=c[2]; u.i[7]=c[3];
  return u.v;
}

// ---------------- Kernel 1: pack x into bf16 WMMA-A tile layout ----------------
// x_pack block (comp i, bb16, mb): 1 KB; lane's 32B contiguous = its 8 A VGPRs.
__global__ __launch_bounds__(512)
void pack_x_kernel(const float* __restrict__ x, unsigned* __restrict__ xp){
  int bb = blockIdx.x;          // 0..NBB-1 (16-row blocks)
  int mb = blockIdx.y;          // 0..NMB-1
  int t    = threadIdx.x;
  int comp = t >> 5;            // wave id == component
  int lane = t & 31;
  int M = lane & 15;
  unsigned d[8];
#pragma unroll
  for (int v=0; v<8; v++){
    int K = ((v<4) ? 2*v : 2*v+8) + ((lane<16) ? 0 : 8);
    size_t base = ((size_t)(bb*16 + M) * F_DIM + (size_t)(mb*32 + K)) * 16 + comp;
    float f0 = x[base];
    float f1 = x[base + 16];    // K+1 -> +16 floats (component-last layout)
    d[v] = pack_bf16(f0, f1);
  }
  unsigned* dst = xp + (((size_t)comp*NBB + bb)*NMB + mb)*256 + (size_t)lane*8;
  *(v4i*)dst       = (v4i){(int)d[0],(int)d[1],(int)d[2],(int)d[3]};
  *(v4i*)(dst + 4) = (v4i){(int)d[4],(int)d[5],(int)d[6],(int)d[7]};
}

// ---------------- Kernel 2: pack w_left/w_right (transposed) into WMMA-B tiles ----
__global__ __launch_bounds__(256)
void pack_w_kernel(const float* __restrict__ wl, const float* __restrict__ wr,
                   unsigned* __restrict__ wlp, unsigned* __restrict__ wrp){
  int gid  = blockIdx.x * 8 + (threadIdx.x >> 5);   // global wave id: 0..5119
  int lane = threadIdx.x & 31;
  int side = gid / (5*NMB*NNB);
  int rem  = gid % (5*NMB*NNB);
  int g  = rem / (NMB*NNB);
  int mb = (rem / NNB) % NMB;
  int nb = rem % NNB;
  const float* w  = side ? wr  : wl;
  unsigned*    wp = side ? wrp : wlp;
  int n = nb*16 + (lane & 15);
  unsigned d[8];
#pragma unroll
  for (int v=0; v<8; v++){
    int K = ((v<4) ? 2*v : 2*v+8) + ((lane<16) ? 0 : 8);
    size_t base = ((size_t)n * F_DIM + (size_t)(mb*32 + K)) * 5 + g;
    float f0 = w[base];
    float f1 = w[base + 5];     // m+1
    d[v] = pack_bf16(f0, f1);
  }
  unsigned* dst = wp + (((size_t)g*NMB + mb)*NNB + nb)*256 + (size_t)lane*8;
  *(v4i*)dst       = (v4i){(int)d[0],(int)d[1],(int)d[2],(int)d[3]};
  *(v4i*)(dst + 4) = (v4i){(int)d[4],(int)d[5],(int)d[6],(int)d[7]};
}

// ---------------- Kernel 3: fused dual-GEMM (32x16 tile) + normalize + gp ------
// WG = 512 threads = 16 waves; wave i owns component i; tile = 32 b x 16 n.
// Two A sub-tiles share one BL/BR pair -> 4 WMMAs per 8 b128 loads.
#define ROWP 18                  // padded LDS row stride (bank-conflict-free)
#define CMPP (16*ROWP)

__global__ __launch_bounds__(512)
void gemm_gp_kernel(const float* __restrict__ x,
                    const float* __restrict__ b_left,
                    const float* __restrict__ a_norm,
                    const float* __restrict__ gp_weight,
                    const unsigned* __restrict__ xp,
                    const unsigned* __restrict__ wlp,
                    const unsigned* __restrict__ wrp,
                    float* __restrict__ out){
  __shared__ float Ls[16*CMPP];   // 16 comps x 16 rows x padded cols (one half)
  __shared__ float Rs[16*CMPP];

  int nb = blockIdx.x;          // 0..NNB-1
  int bb = blockIdx.y;          // 0..NBB2-1  (32-row tiles)
  int t    = threadIdx.x;
  int wave = t >> 5;            // component
  int lane = t & 31;
  int g = GT.grade[wave];

  const unsigned* ap0 = xp  + (((size_t)wave*NBB + bb*2    )*NMB)*256 + (size_t)lane*8;
  const unsigned* ap1 = xp  + (((size_t)wave*NBB + bb*2 + 1)*NMB)*256 + (size_t)lane*8;
  const unsigned* blp = wlp + (((size_t)g*NMB)*NNB + nb)*256          + (size_t)lane*8;
  const unsigned* brp = wrp + (((size_t)g*NMB)*NNB + nb)*256          + (size_t)lane*8;

  v8f accL0 = {0.f,0.f,0.f,0.f,0.f,0.f,0.f,0.f};
  v8f accR0 = {0.f,0.f,0.f,0.f,0.f,0.f,0.f,0.f};
  v8f accL1 = {0.f,0.f,0.f,0.f,0.f,0.f,0.f,0.f};
  v8f accR1 = {0.f,0.f,0.f,0.f,0.f,0.f,0.f,0.f};

#pragma unroll 4
  for (int mb=0; mb<NMB; ++mb){
    v16bf A0 = load_bf16x16(ap0);
    v16bf A1 = load_bf16x16(ap1);
    v16bf BL = load_bf16x16(blp);
    v16bf BR = load_bf16x16(brp);
    accL0 = __builtin_amdgcn_wmma_f32_16x16x32_bf16(false, A0, false, BL,
                                                    (short)0, accL0, false, false);
    accR0 = __builtin_amdgcn_wmma_f32_16x16x32_bf16(false, A0, false, BR,
                                                    (short)0, accR0, false, false);
    accL1 = __builtin_amdgcn_wmma_f32_16x16x32_bf16(false, A1, false, BL,
                                                    (short)0, accL1, false, false);
    accR1 = __builtin_amdgcn_wmma_f32_16x16x32_bf16(false, A1, false, BR,
                                                    (short)0, accR1, false, false);
    ap0 += 256;
    ap1 += 256;
    blp += (size_t)NNB*256;
    brp += (size_t)NNB*256;
  }

  int colw  = lane & 15;
  int rbase = (lane < 16) ? 0 : 8;

#pragma unroll
  for (int half=0; half<2; ++half){
    // ---- store this half's accumulators to LDS ----
    v8f aL = half ? accL1 : accL0;
    v8f aR = half ? accR1 : accR0;
#pragma unroll
    for (int v=0; v<8; v++){
      int idx = wave*CMPP + (v + rbase)*ROWP + colw;
      Ls[idx] = aL[v];
      Rs[idx] = aR[v];
    }
    __syncthreads();

    // ---- epilogue for 16 rows x 16 cols ----
    if (t < 256){
      int row = t >> 4, col = t & 15;
      size_t b = (size_t)bb*32 + (size_t)half*16 + row;
      int n = nb*16 + col;

      float xr[16], R[16], L[16];
      {
        const v4f* xs = (const v4f*)(x + (b*F_DIM + n)*16);
        v4f x0 = xs[0], x1 = xs[1], x2 = xs[2], x3 = xs[3];
#pragma unroll
        for (int i=0;i<4;i++){ xr[i]=x0[i]; xr[4+i]=x1[i]; xr[8+i]=x2[i]; xr[12+i]=x3[i]; }
      }
#pragma unroll
      for (int i=0;i<16;i++){
        int idx = i*CMPP + row*ROWP + col;
        R[i] = Rs[idx];
        L[i] = Ls[idx];
      }

      // per-grade gated normalization of the right branch
      float q[5] = {0.f,0.f,0.f,0.f,0.f};
#pragma unroll
      for (int i=0;i<16;i++) q[GT.grade[i]] += GT.beta[i]*R[i]*R[i];

      float inv[5];
#pragma unroll
      for (int gg=0; gg<5; gg++){
        float nr = sqrtf(fabsf(q[gg]));
        float a  = a_norm[(size_t)n*5 + gg];
        float sg = 1.0f / (1.0f + expf(-a));
        inv[gg]  = 1.0f / (sg*(nr - 1.0f) + 1.0f + 1e-6f);
      }
      float Rn[16];
#pragma unroll
      for (int i=0;i<16;i++) Rn[i] = R[i]*inv[GT.grade[i]];

      float gpw[35];
#pragma unroll
      for (int p=0;p<35;p++) gpw[p] = gp_weight[(size_t)n*35 + p];

      // geometric product: 256 Cayley terms, tables fold to immediates
      float gp[16];
#pragma unroll
      for (int j=0;j<16;j++) gp[j] = 0.f;
#pragma unroll
      for (int i=0;i<16;i++){
#pragma unroll
        for (int k=0;k<16;k++){
          float term = xr[i]*Rn[k];
          float wv = (GT.ss[i][k] < 0.0f) ? -gpw[GT.pp[i][k]] : gpw[GT.pp[i][k]];
          gp[GT.jj[i][k]] = fmaf(wv, term, gp[GT.jj[i][k]]);
        }
      }

      L[0] += b_left[n];
      const float s2 = 0.70710678118654752f;
      v4f o0,o1,o2,o3;
#pragma unroll
      for (int i=0;i<4;i++){
        o0[i] = (L[i]    + gp[i]   )*s2;
        o1[i] = (L[4+i]  + gp[4+i] )*s2;
        o2[i] = (L[8+i]  + gp[8+i] )*s2;
        o3[i] = (L[12+i] + gp[12+i])*s2;
      }
      v4f* od = (v4f*)(out + (b*F_DIM + n)*16);
      od[0]=o0; od[1]=o1; od[2]=o2; od[3]=o3;
    }
    __syncthreads();   // WAR: protect LDS before next half's stores
  }
}

// ---------------- launch ----------------
extern "C" void kernel_launch(void* const* d_in, const int* in_sizes, int n_in,
                              void* d_out, int out_size, void* d_ws, size_t ws_size,
                              hipStream_t stream) {
  const float* x         = (const float*)d_in[0];
  const float* w_left    = (const float*)d_in[1];
  const float* b_left    = (const float*)d_in[2];
  const float* w_right   = (const float*)d_in[3];
  const float* a_norm    = (const float*)d_in[4];
  const float* gp_weight = (const float*)d_in[5];
  float* out = (float*)d_out;

  unsigned char* ws = (unsigned char*)d_ws;
  const size_t XP_BYTES = (size_t)16*NBB*NMB*1024;   // 64 MiB bf16 packed x
  const size_t WP_BYTES = (size_t)5*NMB*NNB*1024;    // 2.5 MiB per weight set
  unsigned* xp  = (unsigned*)ws;
  unsigned* wlp = (unsigned*)(ws + XP_BYTES);
  unsigned* wrp = (unsigned*)(ws + XP_BYTES + WP_BYTES);

  pack_x_kernel<<<dim3(NBB, NMB), 512, 0, stream>>>(x, xp);
  pack_w_kernel<<<(2*5*NMB*NNB)/8, 256, 0, stream>>>(w_left, w_right, wlp, wrp);
  gemm_gp_kernel<<<dim3(NNB, NBB2), 512, 0, stream>>>(x, b_left, a_norm, gp_weight,
                                                      xp, wlp, wrp, out);
}
